// OptimizedSphereConv2D_14336600834202
// MI455X (gfx1250) — matrix operand order repaired and checked by reading, not compiled, not run on hardware
//
#include <hip/hip_runtime.h>

// ---------------------------------------------------------------------------
// Sphere-sample + 3x3 stride-3 conv fused kernel for gfx1250 (MI455X).
// GEMM view: out[m][co] = sum_k A[m][k] * B[k][co],  k = tap*16 + ci (144->160)
// A built per-lane in the native 16-bit A 16x32 WMMA layout (no shuffles).
// bf16 operands, f32 accumulate via v_wmma_f32_16x16x32_bf16.
// ---------------------------------------------------------------------------

typedef __attribute__((ext_vector_type(16))) __bf16 v16bf;
typedef __attribute__((ext_vector_type(8)))  float  v8f;

#define HH    256
#define WW    512
#define CIN   16
#define COUT  32
#define GWID  1536          // grid width  (3*WW)
#define KTOT  144
#define KPAD  160
#define TCOLS 128           // output columns per workgroup (8 waves * 16)

struct Tap {
    int   o00, o01, o10, o11;
    float w00, w01, w10, w11;
};

__device__ __forceinline__ Tap make_tap(const float* __restrict__ grid, int gr, int gc)
{
    const float2 g = *(const float2*)(grid + ((size_t)gr * GWID + gc) * 2);
    // grid_sample, align_corners=True
    float ix = (g.x + 1.0f) * 0.5f * (float)(WW - 1);
    float iy = (g.y + 1.0f) * 0.5f * (float)(HH - 1);
    float fx0 = floorf(ix), fy0 = floorf(iy);
    int x0 = (int)fx0, y0 = (int)fy0;
    int x1 = x0 + 1,  y1 = y0 + 1;
    float wx1 = ix - fx0, wy1 = iy - fy0;
    float wx0 = 1.0f - wx1, wy0 = 1.0f - wy1;
    // zero-padding validity masks folded into the weights
    float vx0 = (x0 >= 0 && x0 < WW) ? 1.0f : 0.0f;
    float vx1 = (x1 >= 0 && x1 < WW) ? 1.0f : 0.0f;
    float vy0 = (y0 >= 0 && y0 < HH) ? 1.0f : 0.0f;
    float vy1 = (y1 >= 0 && y1 < HH) ? 1.0f : 0.0f;
    int cx0 = x0 < 0 ? 0 : (x0 > WW - 1 ? WW - 1 : x0);
    int cx1 = x1 < 0 ? 0 : (x1 > WW - 1 ? WW - 1 : x1);
    int cy0 = y0 < 0 ? 0 : (y0 > HH - 1 ? HH - 1 : y0);
    int cy1 = y1 < 0 ? 0 : (y1 > HH - 1 ? HH - 1 : y1);
    Tap t;
    t.o00 = cy0 * WW + cx0;  t.o01 = cy0 * WW + cx1;
    t.o10 = cy1 * WW + cx0;  t.o11 = cy1 * WW + cx1;
    t.w00 = wx0 * wy0 * vx0 * vy0;
    t.w01 = wx1 * wy0 * vx1 * vy0;
    t.w10 = wx0 * wy1 * vx0 * vy1;
    t.w11 = wx1 * wy1 * vx1 * vy1;
    return t;
}

__device__ __forceinline__ float sample_ch(const float* __restrict__ p, const Tap& t)
{
    return t.w00 * p[t.o00] + t.w01 * p[t.o01] + t.w10 * p[t.o10] + t.w11 * p[t.o11];
}

__global__ __launch_bounds__(256)
void sphere_conv_wmma(const float* __restrict__ x,
                      const float* __restrict__ wgt,
                      const float* __restrict__ bias,
                      const float* __restrict__ grid,
                      float* __restrict__ out)
{
    __shared__ __bf16 sB[COUT][KPAD];     // B[k][co] stored [co][k], bf16, 10 KB
    __shared__ float  sO[COUT][TCOLS];    // output transpose tile, 16 KB

    const int tid  = threadIdx.x;
    const int lane = tid & 31;
    const int wv   = tid >> 5;            // wave id 0..7
    const int b    = blockIdx.z;          // batch 0..3
    const int r    = blockIdx.y;          // output row 0..255
    const int c0wg = blockIdx.x * TCOLS;  // first output column of this WG

    // ---- pack weights (COUT,CIN,3,3) f32 -> bf16 B[co][k], k = tap*16+ci ----
    for (int e = tid; e < COUT * KPAD; e += 256) {
        int co = e / KPAD;
        int k  = e - co * KPAD;
        float v = 0.0f;
        if (k < KTOT)
            v = wgt[(co * CIN + (k & 15)) * 9 + (k >> 4)];   // [co][ci][tap]
        sB[co][k] = (__bf16)v;
    }
    __syncthreads();

    // Lane -> (pixel, ci-half) mapping matching the 16-bit A 16x32 layout:
    //   lanes 0-15  : M = lane, K%32 in {0..7, 16..23}  -> ci 0..7
    //   lanes 16-31 : M = lane-16, K%32 in {8..15,24..31} -> ci 8..15
    const int m    = lane & 15;
    const int cib  = (lane >> 4) << 3;                 // 0 or 8
    const int col  = c0wg + wv * 16 + m;               // this lane's output column
    const float* xb = x + ((size_t)b * CIN + cib) * (size_t)(HH * WW);

    // B fragment addressing: lane<16 -> K {0..7,16..23} of chunk, lane>=16 -> K {8..15,24..31}
    const int nrow = lane & 15;
    const int kofs = (lane >> 4) << 3;                 // 0 or 8

    v8f acc0 = {};   // couts 0..15
    v8f acc1 = {};   // couts 16..31

    #pragma unroll
    for (int kc = 0; kc < 5; ++kc) {                   // K chunks of 32 (160 total)
        v16bf a;
        #pragma unroll
        for (int hlf = 0; hlf < 2; ++hlf) {
            const int tap = 2 * kc + hlf;              // compile-time constant
            if (tap < 9) {
                const int kh = tap / 3, kw = tap % 3;
                Tap t = make_tap(grid, 3 * r + kh, 3 * col + kw);
                #pragma unroll
                for (int ci = 0; ci < 8; ++ci)
                    a[hlf * 8 + ci] =
                        (__bf16)sample_ch(xb + (size_t)ci * (HH * WW), t);
            } else {
                #pragma unroll
                for (int ci = 0; ci < 8; ++ci)
                    a[hlf * 8 + ci] = (__bf16)0.0f;    // K padding 144..159
            }
        }
        v16bf bf0, bf1;
        #pragma unroll
        for (int s = 0; s < 16; ++s) {
            const int kk = kc * 32 + kofs + (s & 7) + ((s >> 3) << 4);
            bf0[s] = sB[nrow][kk];
            bf1[s] = sB[nrow + 16][kk];
        }
        acc0 = __builtin_amdgcn_wmma_f32_16x16x32_bf16(
                   false, a, false, bf0, (short)0, acc0, false, false);
        acc1 = __builtin_amdgcn_wmma_f32_16x16x32_bf16(
                   false, a, false, bf1, (short)0, acc1, false, false);
    }

    // ---- transpose D through LDS for coalesced stores ----
    // C/D layout: lanes 0-15: N=lane, M=reg; lanes 16-31: N=lane-16, M=8+reg.
    {
        const int n = lane & 15;
        #pragma unroll
        for (int rr = 0; rr < 8; ++rr) {
            const int mm = (lane < 16) ? rr : 8 + rr;
            sO[n][wv * 16 + mm]      = acc0[rr];
            sO[n + 16][wv * 16 + mm] = acc1[rr];
        }
    }
    __syncthreads();

    // 32 couts x 128 consecutive columns, 512B-coalesced runs, bias fused.
    for (int e = tid; e < COUT * TCOLS; e += 256) {
        const int co = e >> 7;
        const int cl = e & 127;
        out[(((size_t)b * COUT + co) * HH + r) * WW + (c0wg + cl)] =
            sO[co][cl] + bias[co];
    }
}

// ---------------------------------------------------------------------------

extern "C" void kernel_launch(void* const* d_in, const int* in_sizes, int n_in,
                              void* d_out, int out_size, void* d_ws, size_t ws_size,
                              hipStream_t stream)
{
    (void)in_sizes; (void)n_in; (void)out_size; (void)d_ws; (void)ws_size;
    const float* x    = (const float*)d_in[0];   // (4,16,256,512)
    const float* wgt  = (const float*)d_in[1];   // (32,16,3,3)
    const float* bias = (const float*)d_in[2];   // (32,)
    const float* grid = (const float*)d_in[3];   // (768,1536,2)
    float* out = (float*)d_out;                  // (4,32,256,512)

    dim3 grd(WW / TCOLS, HH, 4);                 // (4, 256, 4)
    dim3 blk(256);                               // 8 waves (wave32)
    sphere_conv_wmma<<<grd, blk, 0, stream>>>(x, wgt, bias, grid, out);
}
